// GraphAttentionalLayer_1168231104632
// MI455X (gfx1250) — matrix operand order repaired and checked by reading, not compiled, not run on hardware
//
#include <hip/hip_runtime.h>
#include <hip/hip_bf16.h>

#define BATCH 8
#define NN    2048
#define DD    128
#define MROWS (BATCH * NN)   // 16384 total rows

typedef __attribute__((ext_vector_type(16))) _Float16 v16h;
typedef __attribute__((ext_vector_type(8)))  _Float16 v8h;
typedef __attribute__((ext_vector_type(8)))  float    v8f;
typedef __attribute__((ext_vector_type(2)))  float    v2f;

// Monotone order-preserving float<->uint mapping for deterministic atomicMax on floats.
__device__ __forceinline__ unsigned order_f32(float x) {
  unsigned u = __float_as_uint(x);
  return (u & 0x80000000u) ? ~u : (u | 0x80000000u);
}
__device__ __forceinline__ float unorder_f32(unsigned u) {
  return __uint_as_float((u & 0x80000000u) ? (u & 0x7FFFFFFFu) : ~u);
}

__global__ void init_kernel(unsigned* __restrict__ smax_u) {
  if (threadIdx.x < BATCH) smax_u[threadIdx.x] = order_f32(-3.0e38f);
}

// Pass 1: hw = h @ W^T in fp32 via V_WMMA_F32_16X16X4_F32.
// Stores hw transposed as f16 (HWt[e][row]) for pass 2, plus s_i, s_j and
// per-batch max_j s_j (atomic, order-preserving uint encoding).
__global__ __launch_bounds__(256) void prep_kernel(
    const float* __restrict__ h, const float* __restrict__ W,
    const float* __restrict__ a, _Float16* __restrict__ hwt,
    float* __restrict__ s_i, float* __restrict__ s_j,
    unsigned* __restrict__ smax_u)
{
  __shared__ float Wl[DD * DD];  // 64 KB: full W staged in LDS
  const int tid = threadIdx.x;
  {
    const float4* Wv = (const float4*)W;
    float4* Lv = (float4*)Wl;
    #pragma unroll
    for (int i = 0; i < 16; ++i) Lv[tid + 256 * i] = Wv[tid + 256 * i];
  }
  __syncthreads();

  const int wave = tid >> 5, lane = tid & 31;
  const int m = lane & 15, hlf = lane >> 4;         // A/B fragment row/col + lane half
  const int rowBase = blockIdx.x * 128 + wave * 16; // 16 rows per wave

  v8f C[8] = {};  // 16x128 fp32 accumulators (8 e-tiles of 16x16)

  // A 16x4 f32 layout: lanes 0-15 -> K pair {k0,k0+1}; lanes 16-31 -> {k0+2,k0+3}
  const float* hrow = h + (size_t)(rowBase + m) * DD + 2 * hlf;
  #pragma unroll 4
  for (int ks = 0; ks < 32; ++ks) {
    v2f av = *(const v2f*)(hrow + 4 * ks);
    #pragma unroll
    for (int e = 0; e < 8; ++e) {
      // B 4x16 f32 (mirrors A): N = e*16 + (lane&15), same K-pair-per-half layout.
      v2f bv = *(const v2f*)(&Wl[(e * 16 + m) * DD + 4 * ks + 2 * hlf]);
      C[e] = __builtin_amdgcn_wmma_f32_16x16x4_f32(false, av, false, bv,
                                                   (short)0, C[e], false, false);
    }
  }

  // s_i/s_j partials + store HWt f16 (transposed: column e, contiguous rows).
  float psi[8] = {}, psj[8] = {};
  #pragma unroll
  for (int e = 0; e < 8; ++e) {
    const int col = e * 16 + m;
    const float ai = a[col], aj = a[DD + col];
    v8h pk;
    #pragma unroll
    for (int k = 0; k < 8; ++k) {
      float v = C[e][k];          // C VGPR k: row = k + 8*hlf, col = e*16 + m
      psi[k] += v * ai;
      psj[k] += v * aj;
      pk[k] = (_Float16)v;
    }
    *(v8h*)(hwt + (size_t)col * MROWS + rowBase + 8 * hlf) = pk;  // 16B store
  }

  // Reduce partials across each 16-lane half (rows k in lanes 0-15, k+8 in 16-31).
  #pragma unroll
  for (int k = 0; k < 8; ++k) {
    #pragma unroll
    for (int mask = 8; mask >= 1; mask >>= 1) {
      psi[k] += __shfl_xor(psi[k], mask, 32);
      psj[k] += __shfl_xor(psj[k], mask, 32);
    }
  }

  if (lane == 0 || lane == 16) {
    const int r0 = rowBase + 8 * hlf;
    float mx = -3.0e38f;
    #pragma unroll
    for (int k = 0; k < 8; ++k) {
      s_i[r0 + k] = psi[k];
      s_j[r0 + k] = psj[k];
      mx = fmaxf(mx, psj[k]);
    }
    atomicMax(&smax_u[rowBase >> 11], order_f32(mx));
  }
}

// Pass 2: fused softmax + P@HW. Row max is exact a priori: m_i = relu(s_i + max_j s_j),
// so a single pass with no online rescale. P in f16, accum f32 via WMMA 16x16x32_f16.
// B operand (32x128 f16 chunk, 8 KB) is staged into LDS once per 8-wave workgroup via
// double-buffered GLOBAL_LOAD_ASYNC_TO_LDS_B128 (ASYNCcnt), overlapping DMA with WMMA.
__global__ __launch_bounds__(256) void attn_kernel(
    const _Float16* __restrict__ hwt, const float* __restrict__ s_i,
    const float* __restrict__ s_j, const unsigned* __restrict__ smax_u,
    float* __restrict__ out)
{
  __shared__ __attribute__((aligned(32))) _Float16 Vl[2][DD * 32]; // 2 x 8 KB chunk buffers

  const int tid = threadIdx.x, wave = tid >> 5, lane = tid & 31;
  const int m = lane & 15, hlf = lane >> 4;
  const int b = blockIdx.x >> 4;               // batch
  const int t = (blockIdx.x & 15) * 8 + wave;  // i-tile within batch (0..127)
  const int row0 = b * NN + t * 16;

  // This thread's 32-byte staging slice: e-row = tid/2, half-row = (tid&1)*16 halfs.
  const int se = tid >> 1;
  const int sh = (tid & 1) * 16;
  const _Float16* gsrc = hwt + (size_t)se * MROWS + b * NN + sh;
  const unsigned lbase = (unsigned)(uintptr_t)&Vl[0][se * 32 + sh];

  const float smax = unorder_f32(smax_u[b]);
  const float siv  = s_i[row0 + m];            // A-fragment row M = lane&15
  const float mi   = fmaxf(siv + smax, 0.0f);  // exact softmax row max
  const float* sj  = s_j + b * NN;

  v8f C[8] = {};      // 16x128 f32 output accumulators
  float lsum = 0.0f;  // per-lane partial softmax denominator
  const int kb = 16 * hlf;

  // Prologue: stage chunk 0 into buffer 0 (async global->LDS, tracked by ASYNCcnt).
  {
    uint64_t g = (uint64_t)(uintptr_t)gsrc;
    asm volatile("global_load_async_to_lds_b128 %0, %1, off\n\t"
                 "global_load_async_to_lds_b128 %0, %1, off offset:16"
                 :: "v"(lbase), "v"(g) : "memory");
    asm volatile("s_wait_asynccnt 0" ::: "memory");
  }
  __syncthreads();

  for (int c = 0; c < 64; ++c) {
    const int j0 = c * 32;
    const int cur = c & 1;

    // Kick off next chunk's DMA into the other buffer while we compute.
    if (c + 1 < 64) {
      uint64_t g = (uint64_t)(uintptr_t)(gsrc + (c + 1) * 32);
      unsigned l = lbase + ((c + 1) & 1) * (unsigned)(DD * 32 * sizeof(_Float16));
      asm volatile("global_load_async_to_lds_b128 %0, %1, off\n\t"
                   "global_load_async_to_lds_b128 %0, %1, off offset:16"
                   :: "v"(l), "v"(g) : "memory");
    }

    // A 16x32 f16 layout: lanes 0-15 hold K 0..15; lanes 16-31 K 16..31
    v16h af;
    #pragma unroll
    for (int q = 0; q < 8; ++q) {
      float p0 = __expf(fmaxf(siv + sj[j0 + kb + q], 0.0f) - mi);
      float p1 = __expf(fmaxf(siv + sj[j0 + kb + 8 + q], 0.0f) - mi);
      lsum += p0 + p1;
      af[q]     = (_Float16)p0;
      af[q + 8] = (_Float16)p1;
    }

    // B 32x16 f16 fragments from LDS: lane N = e*16+m; lanes 0-15 K=0..15, 16-31 K=16..31.
    const _Float16* lsrc = &Vl[cur][m * 32 + 16 * hlf];
    #pragma unroll
    for (int e = 0; e < 8; ++e) {
      v16h bf = *(const v16h*)(lsrc + e * 16 * 32);  // 32B aligned LDS read
      C[e] = __builtin_amdgcn_wmma_f32_16x16x32_f16(false, af, false, bf,
                                                    (short)0, C[e], false, false);
    }

    // Ensure our async loads landed, then sync so (a) next buffer is fully staged
    // by all threads and (b) everyone is done consuming the buffer we overwrite next.
    asm volatile("s_wait_asynccnt 0" ::: "memory");
    __syncthreads();
  }

  // Full row sums: lane L and lane L^16 hold complementary K-groups of row L&15.
  lsum += __shfl_xor(lsum, 16, 32);
  float linv[8];
  #pragma unroll
  for (int k = 0; k < 8; ++k)
    linv[k] = 1.0f / __shfl(lsum, k + 8 * hlf, 32);  // l for row k + 8*hlf

  #pragma unroll
  for (int e = 0; e < 8; ++e) {
    #pragma unroll
    for (int k = 0; k < 8; ++k) {
      const int r = k + 8 * hlf;
      out[(size_t)(row0 + r) * DD + e * 16 + m] = fmaxf(C[e][k] * linv[k], 0.0f);
    }
  }
}

extern "C" void kernel_launch(void* const* d_in, const int* in_sizes, int n_in,
                              void* d_out, int out_size, void* d_ws, size_t ws_size,
                              hipStream_t stream) {
  const float* h = (const float*)d_in[0];   // (B, N, D) f32
  const float* W = (const float*)d_in[1];   // (D, D) f32
  const float* a = (const float*)d_in[2];   // (2D,) f32
  float* out = (float*)d_out;               // (B, N, D) f32

  char* ws = (char*)d_ws;
  _Float16* hwt = (_Float16*)ws;                                   // 4 MiB: HWt[e][row] f16
  float* s_i = (float*)(ws + (size_t)MROWS * DD * sizeof(_Float16));
  float* s_j = s_i + MROWS;
  unsigned* smax = (unsigned*)(s_j + MROWS);

  init_kernel<<<1, 32, 0, stream>>>(smax);
  prep_kernel<<<128, 256, 0, stream>>>(h, W, a, hwt, s_i, s_j, smax);
  attn_kernel<<<128, 256, 0, stream>>>(hwt, s_i, s_j, smax, out);
}
